// ImageLevelContext_53747220742607
// MI455X (gfx1250) — matrix operand order repaired and checked by reading, not compiled, not run on hardware
//
#include <hip/hip_runtime.h>
#include <hip/hip_bf16.h>

typedef __bf16 bf16_t;
typedef __bf16 v8bf  __attribute__((ext_vector_type(8)));
typedef __bf16 v16bf __attribute__((ext_vector_type(16)));
typedef float  v8f   __attribute__((ext_vector_type(8)));

#define N_SP 4096   // H*W
#define CCH  512    // C
#define TCH  256    // transform channels
#define NB   2      // batch

#ifndef USE_ASYNC_LDS
#define USE_ASYNC_LDS 1
#endif

// ---------------------------------------------------------------------------
// CDNA5 async global->LDS copy (16B per lane), tracked by ASYNCcnt
// ---------------------------------------------------------------------------
__device__ __forceinline__ void async_ld16(unsigned ldsoff, const bf16_t* g) {
  asm volatile("global_load_async_to_lds_b128 %0, %1, off"
               :: "v"(ldsoff), "v"((unsigned long long)(uintptr_t)g)
               : "memory");
}
__device__ __forceinline__ unsigned lds_off(const bf16_t* p) {
  return (unsigned)(uintptr_t)p;  // generic LDS ptr: low 32 bits = LDS offset
}

// ---------------------------------------------------------------------------
// x (b,c,n) f32  ->  xbt (b,n,c) bf16   tiled 32x32 transpose
// ---------------------------------------------------------------------------
__global__ void transpose_cvt(const float* __restrict__ x, bf16_t* __restrict__ xbt) {
  __shared__ float tile[32][33];
  int b = blockIdx.z;
  int n0 = blockIdx.x * 32, c0 = blockIdx.y * 32;
  int tx = threadIdx.x, ty = threadIdx.y;  // 32 x 8
  const float* src = x + (size_t)b * CCH * N_SP;
  for (int j = ty; j < 32; j += 8)
    tile[j][tx] = src[(size_t)(c0 + j) * N_SP + n0 + tx];
  __syncthreads();
  bf16_t* dst = xbt + (size_t)b * N_SP * CCH;
  for (int j = ty; j < 32; j += 8)
    dst[(size_t)(n0 + j) * CCH + c0 + tx] = (bf16_t)tile[tx][j];
}

// weight sub-block f32 -> bf16 (row-major rows x cols; src_ld lead, src_off col offset)
__global__ void cvt_w_bf16(const float* __restrict__ src, bf16_t* __restrict__ dst,
                           int rows, int cols, int src_ld, int src_off) {
  int i = blockIdx.x * blockDim.x + threadIdx.x;
  if (i < rows * cols) {
    int r = i / cols, c = i - r * cols;
    dst[i] = (bf16_t)src[(size_t)r * src_ld + src_off + c];
  }
}

// ---------------------------------------------------------------------------
// g_vec[b,c] = mean_n( softmax_n(x[b,c,:]) * x[b,c,:] ) : one block per row
// ---------------------------------------------------------------------------
__global__ void gvec_kernel(const float* __restrict__ x, float* __restrict__ gvec) {
  int row = blockIdx.x;
  const float* p = x + (size_t)row * N_SP;
  int t = threadIdx.x;
  float v[16];
#pragma unroll
  for (int i = 0; i < 16; ++i) v[i] = p[t * 16 + i];

  __shared__ float red[256];
  float m = -3.4e38f;
#pragma unroll
  for (int i = 0; i < 16; ++i) m = fmaxf(m, v[i]);
  red[t] = m; __syncthreads();
  for (int s = 128; s > 0; s >>= 1) {
    if (t < s) red[t] = fmaxf(red[t], red[t + s]);
    __syncthreads();
  }
  m = red[0]; __syncthreads();

  float se = 0.f, sw = 0.f;
#pragma unroll
  for (int i = 0; i < 16; ++i) {
    float e = __expf(v[i] - m);
    se += e; sw += e * v[i];
  }
  red[t] = se; __syncthreads();
  for (int s = 128; s > 0; s >>= 1) {
    if (t < s) red[t] += red[t + s];
    __syncthreads();
  }
  float SE = red[0]; __syncthreads();
  red[t] = sw; __syncthreads();
  for (int s = 128; s > 0; s >>= 1) {
    if (t < s) red[t] += red[t + s];
    __syncthreads();
  }
  if (t == 0) gvec[row] = red[0] / (SE * (float)N_SP);
}

// ---------------------------------------------------------------------------
// bias_k[b,o] = sum_c k_w1[o,c]*g_vec[b,c] (c<512); bias_v likewise from v_w
// ---------------------------------------------------------------------------
__global__ void bias_gemv(const float* __restrict__ k_w1, const float* __restrict__ v_w,
                          const float* __restrict__ gvec,
                          float* __restrict__ bias_k, float* __restrict__ bias_v) {
  int wid = (blockIdx.x * blockDim.x + threadIdx.x) >> 5;
  int lane = threadIdx.x & 31;
  int sel = wid / (NB * TCH);
  int r = wid - sel * NB * TCH;
  int b = r / TCH, o = r - b * TCH;
  const float* w = (sel ? v_w : k_w1) + (size_t)o * (2 * CCH);
  const float* g = gvec + b * CCH;
  float s = 0.f;
  for (int c = lane; c < CCH; c += 32) s += w[c] * g[c];
#pragma unroll
  for (int off = 16; off > 0; off >>= 1) s += __shfl_xor(s, off, 32);
  if (lane == 0) (sel ? bias_v : bias_k)[b * TCH + o] = s;
}

// ---------------------------------------------------------------------------
// in-place row softmax on bf16 matrix rows of length N_SP; one block per row
// ---------------------------------------------------------------------------
__global__ void softmax_bf16(bf16_t* __restrict__ attn) {
  bf16_t* p = attn + (size_t)blockIdx.x * N_SP;
  int t = threadIdx.x;
  float v[16];
#pragma unroll
  for (int i = 0; i < 16; ++i) v[i] = (float)p[t * 16 + i];

  __shared__ float red[256];
  float m = -3.4e38f;
#pragma unroll
  for (int i = 0; i < 16; ++i) m = fmaxf(m, v[i]);
  red[t] = m; __syncthreads();
  for (int s = 128; s > 0; s >>= 1) {
    if (t < s) red[t] = fmaxf(red[t], red[t + s]);
    __syncthreads();
  }
  m = red[0]; __syncthreads();

  float se = 0.f, e[16];
#pragma unroll
  for (int i = 0; i < 16; ++i) { e[i] = __expf(v[i] - m); se += e[i]; }
  red[t] = se; __syncthreads();
  for (int s = 128; s > 0; s >>= 1) {
    if (t < s) red[t] += red[t + s];
    __syncthreads();
  }
  float inv = 1.f / red[0];
#pragma unroll
  for (int i = 0; i < 16; ++i) p[t * 16 + i] = (bf16_t)(e[i] * inv);
}

// ---------------------------------------------------------------------------
// WMMA bf16 GEMM, transpose-free staging:
//   A: row-major MxK (lda)          B: stored NxK (ldb)  [K contiguous rows]
//   C[m][n] = sum_k A[m][k]*B[n][k], epilogue:
//     v = acc*ascale (+ prebias[m]); if gain: v = v*gain[m]+beta[m]; relu opt
//   OUT_TRANS=1: store out[n*ldOut + m] (bf16, 16B/lane contiguous stores)
//   OUT_TRANS=0: store out[m*ldOut + n] (bf16 or f32 scatter)
// Block tile 128x128, BK=32, 8 waves of 32x64; double-buffered LDS fed by
// GLOBAL_LOAD_ASYNC_TO_LDS_B128 (ASYNCcnt), branch-free x2-unrolled pipeline.
// Requires M,N %128==0, K%64==0 (tile count even).
// ---------------------------------------------------------------------------
template <bool OUT_TRANS, bool OUT_F32>
__global__ __launch_bounds__(256) void gemm_wmma_bf16(
    const bf16_t* __restrict__ A, const bf16_t* __restrict__ Bm,
    void* __restrict__ Cout, int K, int lda, int ldb, int ldOut,
    long long sA, long long sB, long long sC,
    const float* __restrict__ gain, const float* __restrict__ beta,
    const float* __restrict__ prebias, int pstride, float ascale, int relu) {
  const int n0 = blockIdx.x * 128;
  const int m0 = blockIdx.y * 128;
  const int bz = blockIdx.z;
  A += (size_t)bz * sA;
  Bm += (size_t)bz * sB;

  const int tid = threadIdx.x, lane = tid & 31, wave = tid >> 5;
  const int wm = (wave & 3) * 32;   // wave M offset
  const int wn = (wave >> 2) * 64;  // wave N offset
  const int half = lane >> 4, l16 = lane & 15;

  __shared__ __align__(64) bf16_t Alds[2][128 * 32];
  __shared__ __align__(64) bf16_t Blds[2][128 * 32];

  // per-thread staging chunks: tile = 128 rows x 32 bf16; 16B chunks
  const int r0 = tid >> 2;              // rows 0..63
  const int r1 = r0 + 64;               // rows 64..127
  const int o8 = (tid & 3) * 8;         // chunk column (elems)
  const bf16_t* gA0 = A + (size_t)(m0 + r0) * lda + o8;
  const bf16_t* gA1 = A + (size_t)(m0 + r1) * lda + o8;
  const bf16_t* gB0 = Bm + (size_t)(n0 + r0) * ldb + o8;
  const bf16_t* gB1 = Bm + (size_t)(n0 + r1) * ldb + o8;
  const unsigned laA0 = lds_off(&Alds[0][r0 * 32 + o8]);
  const unsigned laA1 = lds_off(&Alds[0][r1 * 32 + o8]);
  const unsigned laB0 = lds_off(&Blds[0][r0 * 32 + o8]);
  const unsigned laB1 = lds_off(&Blds[0][r1 * 32 + o8]);

  v8f acc[2][4];
#pragma unroll
  for (int mi = 0; mi < 2; ++mi)
#pragma unroll
    for (int ni = 0; ni < 4; ++ni)
#pragma unroll
      for (int r = 0; r < 8; ++r) acc[mi][ni][r] = 0.0f;

  // one k-step of WMMAs from buffer `buf` (compile-time 0/1 at call sites)
  auto step = [&](int buf) {
    v16bf af[2], bfr[4];
#pragma unroll
    for (int mi = 0; mi < 2; ++mi) {
      const bf16_t* ar = &Alds[buf][(wm + mi * 16 + l16) * 32];
      v8bf lo = *(const v8bf*)(ar + (half ? 16 : 0));   // K 0..7 | 16..23
      v8bf hi = *(const v8bf*)(ar + (half ? 24 : 8));   // K 8..15 | 24..31
      af[mi] = __builtin_shufflevector(lo, hi, 0, 1, 2, 3, 4, 5, 6, 7, 8, 9,
                                       10, 11, 12, 13, 14, 15);
    }
#pragma unroll
    for (int ni = 0; ni < 4; ++ni)
      bfr[ni] = *(const v16bf*)(&Blds[buf][(wn + ni * 16 + l16) * 32 + half * 16]);
#pragma unroll
    for (int mi = 0; mi < 2; ++mi)
#pragma unroll
      for (int ni = 0; ni < 4; ++ni)
        acc[mi][ni] = __builtin_amdgcn_wmma_f32_16x16x32_bf16(
            false, af[mi], false, bfr[ni], (short)0, acc[mi][ni], false, false);
  };

#if USE_ASYNC_LDS
  auto issue = [&](int buf) {
    unsigned bo = buf ? 8192u : 0u;  // 128*32*2 bytes per buffer
    async_ld16(laA0 + bo, gA0);
    async_ld16(laA1 + bo, gA1);
    async_ld16(laB0 + bo, gB0);
    async_ld16(laB1 + bo, gB1);
    gA0 += 32; gA1 += 32; gB0 += 32; gB1 += 32;
  };
  const int nk = K >> 5;  // even by construction (K % 64 == 0)
  issue(0);
  for (int it = 0; it + 2 < nk; it += 2) {
    issue(1);
    asm volatile("s_wait_asynccnt 0x4" ::: "memory");
    __syncthreads();
    step(0);
    __syncthreads();
    issue(0);
    asm volatile("s_wait_asynccnt 0x4" ::: "memory");
    __syncthreads();
    step(1);
    __syncthreads();
  }
  issue(1);
  asm volatile("s_wait_asynccnt 0x4" ::: "memory");
  __syncthreads();
  step(0);
  __syncthreads();
  asm volatile("s_wait_asynccnt 0x0" ::: "memory");
  __syncthreads();
  step(1);
#else
  const int nk = K >> 5;
  for (int it = 0; it < nk; ++it) {
    v8bf tA0 = *(const v8bf*)gA0, tA1 = *(const v8bf*)gA1;
    v8bf tB0 = *(const v8bf*)gB0, tB1 = *(const v8bf*)gB1;
    gA0 += 32; gA1 += 32; gB0 += 32; gB1 += 32;
    __syncthreads();
    *(v8bf*)&Alds[0][r0 * 32 + o8] = tA0;
    *(v8bf*)&Alds[0][r1 * 32 + o8] = tA1;
    *(v8bf*)&Blds[0][r0 * 32 + o8] = tB0;
    *(v8bf*)&Blds[0][r1 * 32 + o8] = tB1;
    __syncthreads();
    step(0);
    __syncthreads();
  }
#endif

  // ---- epilogue ----
#pragma unroll
  for (int mi = 0; mi < 2; ++mi) {
#pragma unroll
    for (int ni = 0; ni < 4; ++ni) {
      const int nn = n0 + wn + ni * 16 + l16;
      const int mb = m0 + wm + mi * 16 + half * 8;
      if (OUT_TRANS) {
        v8bf o;
#pragma unroll
        for (int r = 0; r < 8; ++r) {
          int m = mb + r;
          float v = acc[mi][ni][r] * ascale;
          if (prebias) v += prebias[bz * pstride + m];
          if (gain) v = v * gain[m] + beta[m];
          if (relu) v = fmaxf(v, 0.f);
          o[r] = (bf16_t)v;
        }
        *(v8bf*)((bf16_t*)Cout + (size_t)bz * sC + (size_t)nn * ldOut + mb) = o;
      } else {
#pragma unroll
        for (int r = 0; r < 8; ++r) {
          int m = mb + r;
          float v = acc[mi][ni][r] * ascale;
          if (prebias) v += prebias[bz * pstride + m];
          if (gain) v = v * gain[m] + beta[m];
          if (relu) v = fmaxf(v, 0.f);
          size_t idx = (size_t)bz * sC + (size_t)m * ldOut + nn;
          if (OUT_F32) ((float*)Cout)[idx] = v;
          else         ((bf16_t*)Cout)[idx] = (bf16_t)v;
        }
      }
    }
  }
}

// ---------------------------------------------------------------------------
extern "C" void kernel_launch(void* const* d_in, const int* in_sizes, int n_in,
                              void* d_out, int out_size, void* d_ws, size_t ws_size,
                              hipStream_t stream) {
  const float* x    = (const float*)d_in[0];
  const float* q_w1 = (const float*)d_in[1];
  const float* q_g1 = (const float*)d_in[2];
  const float* q_b1 = (const float*)d_in[3];
  const float* q_w2 = (const float*)d_in[4];
  const float* q_g2 = (const float*)d_in[5];
  const float* q_b2 = (const float*)d_in[6];
  const float* k_w1 = (const float*)d_in[7];
  const float* k_g1 = (const float*)d_in[8];
  const float* k_b1 = (const float*)d_in[9];
  const float* k_w2 = (const float*)d_in[10];
  const float* k_g2 = (const float*)d_in[11];
  const float* k_b2 = (const float*)d_in[12];
  const float* v_w  = (const float*)d_in[13];
  const float* v_g  = (const float*)d_in[14];
  const float* v_b  = (const float*)d_in[15];
  const float* o_w  = (const float*)d_in[16];
  const float* o_g  = (const float*)d_in[17];
  const float* o_b  = (const float*)d_in[18];
  float* out = (float*)d_out;

  char* ws = (char*)d_ws;
  size_t off = 0;
  auto alloc = [&](size_t bytes) -> void* {
    void* p = ws + off;
    off += (bytes + 255) & ~(size_t)255;
    return p;
  };
  bf16_t* xbt  = (bf16_t*)alloc((size_t)NB * N_SP * CCH * 2);  // [b][n][c]
  bf16_t* q1b  = (bf16_t*)alloc((size_t)NB * N_SP * TCH * 2);  // [b][n][tc]
  bf16_t* k1b  = (bf16_t*)alloc((size_t)NB * N_SP * TCH * 2);  // [b][n][tc]
  bf16_t* qb   = (bf16_t*)alloc((size_t)NB * N_SP * TCH * 2);  // [b][n][tc]
  bf16_t* kb   = (bf16_t*)alloc((size_t)NB * N_SP * TCH * 2);  // [b][m][tc]
  bf16_t* vb   = (bf16_t*)alloc((size_t)NB * TCH * N_SP * 2);  // [b][tc][m]
  bf16_t* ctxb = (bf16_t*)alloc((size_t)NB * N_SP * TCH * 2);  // [b][n][tc]
  bf16_t* attn = (bf16_t*)alloc((size_t)NB * N_SP * N_SP * 2); // [b][n][m]
  bf16_t* wq1b = (bf16_t*)alloc((size_t)TCH * CCH * 2);
  bf16_t* wq2b = (bf16_t*)alloc((size_t)TCH * TCH * 2);
  bf16_t* wk1b = (bf16_t*)alloc((size_t)TCH * CCH * 2);
  bf16_t* wk2b = (bf16_t*)alloc((size_t)TCH * TCH * 2);
  bf16_t* wvxb = (bf16_t*)alloc((size_t)TCH * CCH * 2);
  bf16_t* owb  = (bf16_t*)alloc((size_t)CCH * TCH * 2);
  float* gvec   = (float*)alloc((size_t)NB * CCH * 4);
  float* bias_k = (float*)alloc((size_t)NB * TCH * 4);
  float* bias_v = (float*)alloc((size_t)NB * TCH * 4);

  // ---- packing ----
  transpose_cvt<<<dim3(N_SP / 32, CCH / 32, NB), dim3(32, 8), 0, stream>>>(x, xbt);
  cvt_w_bf16<<<(TCH * CCH + 255) / 256, 256, 0, stream>>>(q_w1, wq1b, TCH, CCH, CCH, 0);
  cvt_w_bf16<<<(TCH * TCH + 255) / 256, 256, 0, stream>>>(q_w2, wq2b, TCH, TCH, TCH, 0);
  cvt_w_bf16<<<(TCH * CCH + 255) / 256, 256, 0, stream>>>(k_w1, wk1b, TCH, CCH, 2 * CCH, CCH);
  cvt_w_bf16<<<(TCH * TCH + 255) / 256, 256, 0, stream>>>(k_w2, wk2b, TCH, TCH, TCH, 0);
  cvt_w_bf16<<<(TCH * CCH + 255) / 256, 256, 0, stream>>>(v_w, wvxb, TCH, CCH, 2 * CCH, CCH);
  cvt_w_bf16<<<(CCH * TCH + 255) / 256, 256, 0, stream>>>(o_w, owb, CCH, TCH, TCH, 0);

  gvec_kernel<<<NB * CCH, 256, 0, stream>>>(x, gvec);
  bias_gemv<<<(2 * NB * TCH * 32) / 256, 256, 0, stream>>>(k_w1, v_w, gvec, bias_k, bias_v);

  dim3 blk(256);
  const long long sX = (long long)N_SP * CCH;
  const long long sT = (long long)N_SP * TCH;
  const long long sS = (long long)N_SP * N_SP;

  // q1[n][o] = relu(g1*(q_w1 . x[n]) + b1)
  gemm_wmma_bf16<true, false><<<dim3(N_SP / 128, TCH / 128, NB), blk, 0, stream>>>(
      wq1b, xbt, q1b, CCH, CCH, CCH, TCH, 0, sX, sT, q_g1, q_b1, nullptr, 0, 1.f, 1);
  // k1[n][o] = relu(g1*(k_w1x . x[n] + bias_k) + b1)
  gemm_wmma_bf16<true, false><<<dim3(N_SP / 128, TCH / 128, NB), blk, 0, stream>>>(
      wk1b, xbt, k1b, CCH, CCH, CCH, TCH, 0, sX, sT, k_g1, k_b1, bias_k, TCH, 1.f, 1);
  // v[c][m] = relu(g*(v_wx . x[m] + bias_v) + b)   (classic layout)
  gemm_wmma_bf16<false, false><<<dim3(N_SP / 128, TCH / 128, NB), blk, 0, stream>>>(
      wvxb, xbt, vb, CCH, CCH, CCH, N_SP, 0, sX, sT, v_g, v_b, bias_v, TCH, 1.f, 1);
  // q[n][o] = relu(g2*(q_w2 . q1[n]) + b2)
  gemm_wmma_bf16<true, false><<<dim3(N_SP / 128, TCH / 128, NB), blk, 0, stream>>>(
      wq2b, q1b, qb, TCH, TCH, TCH, TCH, 0, sT, sT, q_g2, q_b2, nullptr, 0, 1.f, 1);
  // k[m][o] = relu(g2*(k_w2 . k1[m]) + b2)
  gemm_wmma_bf16<true, false><<<dim3(N_SP / 128, TCH / 128, NB), blk, 0, stream>>>(
      wk2b, k1b, kb, TCH, TCH, TCH, TCH, 0, sT, sT, k_g2, k_b2, nullptr, 0, 1.f, 1);
  // sim: A=k[m][c], B=q[n][c] -> attn[n][m] = (1/16) k[m].q[n]   (trans epi)
  gemm_wmma_bf16<true, false><<<dim3(N_SP / 128, N_SP / 128, NB), blk, 0, stream>>>(
      kb, qb, attn, TCH, TCH, TCH, N_SP, sT, sT, sS, nullptr, nullptr, nullptr, 0, 0.0625f, 0);
  // softmax rows over m
  softmax_bf16<<<NB * N_SP, 256, 0, stream>>>(attn);
  // ctx: A=v[c][m], B=attn[n][m] -> ctx[n][c] = sum_m v[c,m] attn[n,m]
  gemm_wmma_bf16<true, false><<<dim3(N_SP / 128, TCH / 128, NB), blk, 0, stream>>>(
      vb, attn, ctxb, N_SP, N_SP, N_SP, TCH, sT, sS, sT, nullptr, nullptr, nullptr, 0, 1.f, 0);
  // out: A=o_w[co][c], B=ctx[n][c] -> d_out[co][n] fp32 classic
  gemm_wmma_bf16<false, true><<<dim3(N_SP / 128, CCH / 128, NB), blk, 0, stream>>>(
      owb, ctxb, out, TCH, TCH, TCH, N_SP, 0, sT, (long long)CCH * N_SP,
      o_g, o_b, nullptr, 0, 1.f, 1);
}
